// MultiHeadAttention_23373212025114
// MI455X (gfx1250) — compile-verified, matrix operand order checked by reading, not run on hardware
//
#include <hip/hip_runtime.h>
#include <hip/hip_bf16.h>

#define BB 4
#define SS 2048
#define DD 768
#define HH 12
#define KD 64
#define ODIM 768

typedef _Float16 h16;
typedef __attribute__((ext_vector_type(16))) _Float16 v16h;
typedef __attribute__((ext_vector_type(8)))  _Float16 v8h;
typedef __attribute__((ext_vector_type(4)))  _Float16 v4h;
typedef __attribute__((ext_vector_type(8)))  float    v8f;

union F16F {
    v16h v;
    v8h  h[2];
};

// Load one 16x16x32 WMMA A/B fragment (f16). rowbase points at the start of
// the 32-element K window for this lane's row (A: row M = lane%16, B: row N).
// Per ISA layout: lane group g holds K = [8g, 8g+8) and [16+8g, 16+8g+8).
__device__ __forceinline__ v16h load_frag(const h16* rowbase, int g) {
    F16F f;
    f.h[0] = *(const v8h*)(rowbase + 8 * g);
    f.h[1] = *(const v8h*)(rowbase + 16 + 8 * g);
    return f.v;
}

__device__ __forceinline__ v8f wmma_f16(v16h a, v16h b, v8f c) {
    return __builtin_amdgcn_wmma_f32_16x16x32_f16(
        /*neg_a=*/false, a, /*neg_b=*/false, b,
        /*c_mod=*/(short)0, c, /*reuse_a=*/false, /*reuse_b=*/false);
}

// CDNA5 async DMA: copy 16B from global memory directly into LDS, tracked by
// ASYNCcnt (no VGPR round-trip). LDS destination is the wave-relative byte
// offset = low 32 bits of the generic shared-memory pointer.
__device__ __forceinline__ void async_b128(void* lds_ptr, const void* gptr) {
    uint32_t lds_off = (uint32_t)(uintptr_t)lds_ptr;
    uint64_t gaddr   = (uint64_t)(uintptr_t)gptr;
    asm volatile("global_load_async_to_lds_b128 %0, %1, off"
                 :: "v"(lds_off), "v"(gaddr) : "memory");
}

__device__ __forceinline__ void wait_async0() {
    asm volatile("s_wait_asynccnt 0" ::: "memory");
}

// ---------------------------------------------------------------------------
// Kernel 0a: one-time weight transpose + fp32->f16 convert: WT[n][k] = W[k][n].
// ---------------------------------------------------------------------------
__global__ __launch_bounds__(256) void wt_kernel(
    const float* __restrict__ W, h16* __restrict__ WT)
{
    __shared__ float tile[32][33];
    const int k0 = blockIdx.x * 32, n0 = blockIdx.y * 32;
    const int tx = threadIdx.x & 31, ty = threadIdx.x >> 5;   // 32 x 8
    #pragma unroll
    for (int i = 0; i < 4; ++i) {
        int k = ty + i * 8;
        tile[k][tx] = W[(size_t)(k0 + k) * ODIM + n0 + tx];
    }
    __syncthreads();
    #pragma unroll
    for (int i = 0; i < 4; ++i) {
        int n = ty + i * 8;
        WT[(size_t)(n0 + n) * DD + k0 + tx] = (h16)tile[tx][n];
    }
}

// ---------------------------------------------------------------------------
// Kernel 0b: elementwise fp32 -> f16 (vectorized float4 -> 4x f16).
// ---------------------------------------------------------------------------
__global__ __launch_bounds__(256) void cvt_kernel(
    const float* __restrict__ X, h16* __restrict__ Xh)
{
    int i = blockIdx.x * 256 + threadIdx.x;
    float4 f = ((const float4*)X)[i];
    v4h hv; hv[0] = (h16)f.x; hv[1] = (h16)f.y;
            hv[2] = (h16)f.z; hv[3] = (h16)f.w;
    ((v4h*)Xh)[i] = hv;
}

// ---------------------------------------------------------------------------
// Kernel 1: f16 GEMM, C[M,N] = Ah[M,768] * WT^T + bias. Block tile 128x64,
// 8 waves 4x2, 32x32 per wave (4 WMMA / k-step). Double-buffered LDS with
// global_load_async_to_lds_b128: tile t+1 DMA overlaps tile t WMMAs.
// mode 0: f16 out[((b*H+h)*S + s)*64 + kd]   (Q, K layout)
// mode 2: f16 out[((b*H+h)*64 + kd)*S + s]   (V transposed)
// mode 3: f32 out[m][n] = (acc + bias) * q_mask[m]
// ---------------------------------------------------------------------------
__global__ __launch_bounds__(256) void gemm_kernel(
    const h16* __restrict__ Ah, const h16* __restrict__ WT,
    const float* __restrict__ bias, const int* __restrict__ qmask,
    h16* __restrict__ outh, float* __restrict__ outf, int mode)
{
    __shared__ __align__(16) h16 sA[2][128][32];   // 16 KB ping-pong
    __shared__ __align__(16) h16 sB[2][64][32];    //  8 KB ping-pong

    const int tid = threadIdx.x;
    const int lane = tid & 31, w = tid >> 5;
    const int ln = lane & 15, g = lane >> 4;
    const int wm = w & 3, wn = w >> 2;          // 4 x 2 wave grid
    const int m0 = blockIdx.y * 128;
    const int n0 = blockIdx.x * 64;

    v8f acc00 = {}, acc01 = {}, acc10 = {}, acc11 = {};

    const int sRow = tid >> 2, sCol = (tid & 3) * 8;   // 64 rows x 32 cols/pass

    // per-thread source pointers (advance by 32 halves per k-step)
    const h16* pA0 = &Ah[(size_t)(m0 + sRow) * DD + sCol];
    const h16* pA1 = &Ah[(size_t)(m0 + sRow + 64) * DD + sCol];
    const h16* pB  = &WT[(size_t)(n0 + sRow) * DD + sCol];

    // prologue: DMA tile 0 into buffer 0
    async_b128(&sA[0][sRow][sCol], pA0);
    async_b128(&sA[0][sRow + 64][sCol], pA1);
    async_b128(&sB[0][sRow][sCol], pB);

    int buf = 0;
    for (int k0 = 0; k0 < DD; k0 += 32) {
        wait_async0();          // tile t landed (t+1 not yet issued)
        __syncthreads();        // visible to all waves

        if (k0 + 32 < DD) {     // DMA tile t+1 into the other buffer,
            int nb = buf ^ 1;   // overlapping the WMMAs below
            async_b128(&sA[nb][sRow][sCol],      pA0 + k0 + 32);
            async_b128(&sA[nb][sRow + 64][sCol], pA1 + k0 + 32);
            async_b128(&sB[nb][sRow][sCol],      pB  + k0 + 32);
        }

        v16h a0 = load_frag(&sA[buf][wm * 32 + ln][0], g);
        v16h a1 = load_frag(&sA[buf][wm * 32 + 16 + ln][0], g);
        v16h b0 = load_frag(&sB[buf][wn * 32 + ln][0], g);
        v16h b1 = load_frag(&sB[buf][wn * 32 + 16 + ln][0], g);
        acc00 = wmma_f16(a0, b0, acc00);
        acc01 = wmma_f16(a0, b1, acc01);
        acc10 = wmma_f16(a1, b0, acc10);
        acc11 = wmma_f16(a1, b1, acc11);

        __syncthreads();        // all reads of buf done before t+2 overwrites
        buf ^= 1;
    }

    #pragma unroll
    for (int mi = 0; mi < 2; ++mi) {
        v8f* row0 = mi ? &acc10 : &acc00;
        v8f* row1 = mi ? &acc11 : &acc01;
        #pragma unroll
        for (int r = 0; r < 8; ++r) {
            int m = m0 + wm * 32 + mi * 16 + r + 8 * g;
            int bidx = m / SS, s = m % SS;
            #pragma unroll
            for (int ni = 0; ni < 2; ++ni) {
                int gn = n0 + wn * 32 + ni * 16 + ln;
                float val = (ni ? (*row1)[r] : (*row0)[r]) + bias[gn];
                if (mode == 3) {
                    float qm = qmask[m] ? 1.0f : 0.0f;
                    outf[(size_t)m * ODIM + gn] = val * qm;
                } else {
                    int hh = gn >> 6, kd = gn & 63;
                    if (mode == 2)
                        outh[((size_t)(bidx * HH + hh) * KD + kd) * SS + s] = (h16)val;
                    else
                        outh[((size_t)(bidx * HH + hh) * SS + s) * KD + kd] = (h16)val;
                }
            }
        }
    }
}

// ---------------------------------------------------------------------------
// Kernel 2: causal flash attention. Each wave owns 16 query rows of one
// (b, h); iterates 32-key tiles with online softmax.
// ---------------------------------------------------------------------------
__global__ __launch_bounds__(256) void attn_kernel(
    const h16* __restrict__ Qp, const h16* __restrict__ Kp,
    const h16* __restrict__ Vt, const int* __restrict__ vmask,
    h16* __restrict__ attn)
{
    __shared__ __align__(16) h16 sP[8][16][32];   // per-wave P transpose buf

    const int tid = threadIdx.x;
    const int lane = tid & 31, w = tid >> 5;
    const int ln = lane & 15, g = lane >> 4;
    const int bh = blockIdx.x;
    const int b = bh / HH, hh = bh % HH;
    const int qbase = blockIdx.y * 128 + w * 16;

    // Q fragments (K-dim 64 -> two 16x32 A fragments), loaded once
    const h16* qptr = Qp + ((size_t)bh * SS + qbase + ln) * KD;
    v16h aq0 = load_frag(qptr, g);
    v16h aq1 = load_frag(qptr + 32, g);

    float mrow[8], lrow[8];
    v8f o0 = {}, o1 = {}, o2 = {}, o3 = {};
    #pragma unroll
    for (int r = 0; r < 8; ++r) { mrow[r] = -3.0e38f; lrow[r] = 0.f; }

    const int kmax = qbase + 16;                // causal bound (exclusive)
    for (int kb = 0; kb < kmax; kb += 32) {
        // S tile 16x32 = two 16x16 C fragments; K-dim 64 -> 2 WMMA each
        const h16* kp0 = Kp + ((size_t)bh * SS + kb + ln) * KD;
        const h16* kp1 = Kp + ((size_t)bh * SS + kb + 16 + ln) * KD;
        v8f s0 = {}, s1 = {};
        s0 = wmma_f16(aq0, load_frag(kp0, g), s0);
        s0 = wmma_f16(aq1, load_frag(kp0 + 32, g), s0);
        s1 = wmma_f16(aq0, load_frag(kp1, g), s1);
        s1 = wmma_f16(aq1, load_frag(kp1 + 32, g), s1);

        const int key0 = kb + ln;
        const int key1 = kb + 16 + ln;
        const int vm0 = vmask[b * SS + key0];
        const int vm1 = vmask[b * SS + key1];

        float p0[8], p1[8];
        #pragma unroll
        for (int r = 0; r < 8; ++r) {
            int qi = qbase + r + 8 * g;
            float x0 = s0[r] * 0.125f;          // 1/sqrt(64)
            float x1 = s1[r] * 0.125f;
            if (key0 > qi || !vm0) x0 = -3.0e38f;
            if (key1 > qi || !vm1) x1 = -3.0e38f;
            // row max across the 16 lanes holding this row's 32 columns
            float v = fmaxf(x0, x1);
            #pragma unroll
            for (int off = 8; off > 0; off >>= 1)
                v = fmaxf(v, __shfl_xor(v, off, 16));
            float mn = fmaxf(mrow[r], v);
            float sc = __expf(mrow[r] - mn);
            mrow[r] = mn;
            float e0 = __expf(x0 - mn);
            float e1 = __expf(x1 - mn);
            float rs = e0 + e1;
            #pragma unroll
            for (int off = 8; off > 0; off >>= 1)
                rs += __shfl_xor(rs, off, 16);
            lrow[r] = lrow[r] * sc + rs;
            o0[r] *= sc; o1[r] *= sc; o2[r] *= sc; o3[r] *= sc;
            p0[r] = e0; p1[r] = e1;
        }

        // Transpose P (C layout) -> A layout via per-wave LDS
        #pragma unroll
        for (int r = 0; r < 8; ++r) {
            sP[w][r + 8 * g][ln]      = (h16)p0[r];
            sP[w][r + 8 * g][16 + ln] = (h16)p1[r];
        }
        asm volatile("s_wait_dscnt 0" ::: "memory");
        v16h aP = load_frag(&sP[w][ln][0], g);

        // O(16x64) += P(16x32) @ V(32x64): 4 WMMAs, B from transposed Vt
        const h16* vb = Vt + ((size_t)bh * KD + ln) * SS + kb;
        o0 = wmma_f16(aP, load_frag(vb, g), o0);
        o1 = wmma_f16(aP, load_frag(vb + 16 * SS, g), o1);
        o2 = wmma_f16(aP, load_frag(vb + 32 * SS, g), o2);
        o3 = wmma_f16(aP, load_frag(vb + 48 * SS, g), o3);
    }

    #pragma unroll
    for (int r = 0; r < 8; ++r) {
        int qi = qbase + r + 8 * g;
        float inv = lrow[r] > 0.f ? 1.0f / lrow[r] : 0.f;
        size_t base = (size_t)(b * SS + qi) * ODIM + hh * KD;
        attn[base + 0 * 16 + ln] = (h16)(o0[r] * inv);
        attn[base + 1 * 16 + ln] = (h16)(o1[r] * inv);
        attn[base + 2 * 16 + ln] = (h16)(o2[r] * inv);
        attn[base + 3 * 16 + ln] = (h16)(o3[r] * inv);
    }
}

// ---------------------------------------------------------------------------
extern "C" void kernel_launch(void* const* d_in, const int* in_sizes, int n_in,
                              void* d_out, int out_size, void* d_ws, size_t ws_size,
                              hipStream_t stream) {
    const float* q      = (const float*)d_in[0];
    const float* k      = (const float*)d_in[1];
    const float* v      = (const float*)d_in[2];
    const int*   q_mask = (const int*)d_in[3];
    const int*   v_mask = (const int*)d_in[4];
    const float* Wq     = (const float*)d_in[5];
    const float* bq     = (const float*)d_in[6];
    const float* Wk     = (const float*)d_in[7];
    const float* bk     = (const float*)d_in[8];
    const float* Wv     = (const float*)d_in[9];
    const float* bv     = (const float*)d_in[10];
    const float* Wo     = (const float*)d_in[11];
    const float* bo     = (const float*)d_in[12];

    const size_t projElems = (size_t)BB * HH * SS * KD;   // 6.29M f16 each
    const size_t wtElems   = (size_t)DD * ODIM;           // 589824 f16 each
    h16* Qp   = (h16*)d_ws;
    h16* Kp   = Qp + projElems;
    h16* Vt   = Kp + projElems;
    h16* attn = Vt + projElems;
    h16* Xh   = attn + projElems;                          // reused for q,k,v
    h16* WTq  = Xh + projElems;
    h16* WTk  = WTq + wtElems;
    h16* WTv  = WTk + wtElems;
    h16* WTo  = WTv + wtElems;

    dim3 gwt(DD / 32, ODIM / 32);                          // 24 x 24
    wt_kernel<<<gwt, 256, 0, stream>>>(Wq, WTq);
    wt_kernel<<<gwt, 256, 0, stream>>>(Wk, WTk);
    wt_kernel<<<gwt, 256, 0, stream>>>(Wv, WTv);
    wt_kernel<<<gwt, 256, 0, stream>>>(Wo, WTo);

    const int nCvtBlk = (int)(projElems / 4 / 256);        // 6144
    dim3 gblk(ODIM / 64, (BB * SS) / 128);                 // 12 x 64

    cvt_kernel<<<nCvtBlk, 256, 0, stream>>>(q, Xh);
    gemm_kernel<<<gblk, 256, 0, stream>>>(Xh, WTq, bq, nullptr, Qp, nullptr, 0);
    cvt_kernel<<<nCvtBlk, 256, 0, stream>>>(k, Xh);
    gemm_kernel<<<gblk, 256, 0, stream>>>(Xh, WTk, bk, nullptr, Kp, nullptr, 0);
    cvt_kernel<<<nCvtBlk, 256, 0, stream>>>(v, Xh);
    gemm_kernel<<<gblk, 256, 0, stream>>>(Xh, WTv, bv, nullptr, Vt, nullptr, 2);

    attn_kernel<<<dim3(BB * HH, SS / 128), 256, 0, stream>>>(Qp, Kp, Vt, v_mask, attn);

    gemm_kernel<<<gblk, 256, 0, stream>>>(attn, WTo, bo, q_mask, nullptr, (float*)d_out, 3);
}
